// FeatureAttention_39127152066811
// MI455X (gfx1250) — matrix operand order repaired and checked by reading, not compiled, not run on hardware
//
#include <hip/hip_runtime.h>
#include <hip/hip_bf16.h>
#include <math.h>

typedef __attribute__((ext_vector_type(16))) _Float16 v16h;
typedef __attribute__((ext_vector_type(8)))  float    v8f;
typedef __attribute__((ext_vector_type(4)))  float    v4f;

#define C_DIM 256
#define R_DIM 64
#define G_DIM 4096

static constexpr int ZSTRIDE = C_DIM + 8;  // padded LDS row stride (halfs)
static constexpr int HSTRIDE = R_DIM + 8;

// ---------------------------------------------------------------------------
// Kernel 0: convert w1 [R,C] and w2 [C,R] from f32 to f16 (one-time, tiny)
// ---------------------------------------------------------------------------
__global__ __launch_bounds__(256) void convert_w_kernel(
    const float* __restrict__ w1, const float* __restrict__ w2,
    _Float16* __restrict__ w1h, _Float16* __restrict__ w2h)
{
    int i = blockIdx.x * 256 + threadIdx.x;          // 0 .. 32767
    if (i < R_DIM * C_DIM) {
        w1h[i] = (_Float16)w1[i];
    } else {
        int j = i - R_DIM * C_DIM;
        w2h[j] = (_Float16)w2[j];
    }
}

// ---------------------------------------------------------------------------
// Kernel 1: segment max + sum pooling. One block per graph, thread = channel.
// batch is sorted, so binary-search the segment bounds (deterministic, no
// atomics). Row loop unrolled x4 for MLP against HBM latency; speculative
// prefetch two unroll-groups ahead (global_prefetch_b8, safely dropped OOB).
// ---------------------------------------------------------------------------
__global__ __launch_bounds__(256) void segpool_kernel(
    const float* __restrict__ x, const int* __restrict__ batch, int N,
    float* __restrict__ segmax, float* __restrict__ segsum)
{
    const int g = blockIdx.x;

    int lo = 0, hi = N;
    while (lo < hi) { int mid = (lo + hi) >> 1; if (batch[mid] < g)     lo = mid + 1; else hi = mid; }
    const int start = lo;
    hi = N;
    while (lo < hi) { int mid = (lo + hi) >> 1; if (batch[mid] < g + 1) lo = mid + 1; else hi = mid; }
    const int end = lo;

    const int c = threadIdx.x;
    float mx = -INFINITY, sm = 0.f;

    int r = start;
    for (; r + 4 <= end; r += 4) {
        __builtin_prefetch(&x[(size_t)(r + 8) * C_DIM + c], 0, 0);
        float v0 = x[(size_t)(r + 0) * C_DIM + c];
        float v1 = x[(size_t)(r + 1) * C_DIM + c];
        float v2 = x[(size_t)(r + 2) * C_DIM + c];
        float v3 = x[(size_t)(r + 3) * C_DIM + c];
        mx = fmaxf(mx, fmaxf(fmaxf(v0, v1), fmaxf(v2, v3)));
        sm += (v0 + v1) + (v2 + v3);
    }
    for (; r < end; ++r) {
        float v = x[(size_t)r * C_DIM + c];
        mx = fmaxf(mx, v);
        sm += v;
    }

    if (start == end) mx = 0.f;   // torch_scatter fills empty 'max' segments with 0
    segmax[(size_t)g * C_DIM + c] = mx;
    segsum[(size_t)g * C_DIM + c] = sm;
}

// ---------------------------------------------------------------------------
// Kernel 2: fused MLP via WMMA.
//   h_p = relu(z_p @ w1^T)  [16,64] per pool p in {max,sum}
//   y   = sigmoid(h_max @ w2^T + h_sum @ w2^T)   [16,256]
// One workgroup (8 waves) per 16-row slab of G.
// Fragment layouts: A, B use contiguous K per lane half (k = 16*(lane/16) + s); C/D per CDNA5 ISA 7.12.2.
// ---------------------------------------------------------------------------
__device__ __forceinline__ v16h load_a_frag(const _Float16* __restrict__ base,
                                            int stride, int m, int khi, int k0)
{
    // A 16x32 f16: lanes 0-15 (khi=0): K 0..15 contiguous
    //              lanes 16-31 (khi=8): K 16..31 contiguous
    v16h a;
#pragma unroll
    for (int h = 0; h < 16; ++h) {
        int k = 2 * khi + h;
        a[h] = base[m * stride + k0 + k];
    }
    return a;
}

__device__ __forceinline__ v16h load_b_frag_wT(const _Float16* __restrict__ w,
                                               int rowStride, int n, int kb16,
                                               int n0, int k0)
{
    // B 32x16 f16, B[k][nn] = w[n0+nn][k]: lane group 0 -> K k0+0..15,
    // lane group 1 -> K k0+16..31; 16 consecutive halfs per lane.
    v16h b;
#pragma unroll
    for (int h = 0; h < 16; ++h) {
        b[h] = w[(n0 + n) * rowStride + (k0 + kb16 + h)];
    }
    return b;
}

__global__ __launch_bounds__(256) void mlp_wmma_kernel(
    const float* __restrict__ segmax, const float* __restrict__ segsum,
    const _Float16* __restrict__ w1h, const _Float16* __restrict__ w2h,
    float* __restrict__ y)
{
    __shared__ _Float16 zs[2][16 * ZSTRIDE];   // pooled slabs (f16)
    __shared__ _Float16 hs[2][16 * HSTRIDE];   // relu(hidden) (f16)

    const int tid  = threadIdx.x;
    const int lane = tid & 31;
    const int wave = tid >> 5;
    const int g0   = blockIdx.x * 16;

    // ---- stage both pooled slabs [16 x 256] into LDS as f16 (coalesced) ----
#pragma unroll
    for (int j = 0; j < 16; ++j) {
        int idx = tid + j * 256;               // 0..4095
        int row = idx >> 8, col = idx & 255;
        zs[0][row * ZSTRIDE + col] = (_Float16)segmax[(size_t)g0 * C_DIM + idx];
        zs[1][row * ZSTRIDE + col] = (_Float16)segsum[(size_t)g0 * C_DIM + idx];
    }
    __syncthreads();

    const int m    = lane & 15;
    const int khi  = (lane >> 4) * 8;
    const int nn   = lane & 15;
    const int kb16 = (lane >> 4) * 16;
    const int mb   = (lane >> 4) * 8;

    // ---- GEMM1: wave w -> pool = w>>2, hidden col tile n0 = (w&3)*16 ----
    {
        const int pool = wave >> 2;
        const int n0   = (wave & 3) * 16;
        v8f acc = {};
#pragma unroll
        for (int kb = 0; kb < C_DIM / 32; ++kb) {
            const int k0 = kb * 32;
            v16h a = load_a_frag(zs[pool], ZSTRIDE, m, khi, k0);
            v16h b = load_b_frag_wT(w1h, C_DIM, nn, kb16, n0, k0);
            acc = __builtin_amdgcn_wmma_f32_16x16x32_f16(
                false, a, false, b, (short)0, acc, false, false);
        }
#pragma unroll
        for (int r = 0; r < 8; ++r) {
            float v = acc[r];
            hs[pool][(mb + r) * HSTRIDE + n0 + nn] = (_Float16)(v > 0.f ? v : 0.f);
        }
    }
    __syncthreads();

    // ---- GEMM2: output col tiles; wave handles n0 = wave*16 and (wave+8)*16.
    //      Accumulate both pools into one accumulator (the '+' is free). ----
#pragma unroll
    for (int t = 0; t < 2; ++t) {
        const int n0 = (wave + t * 8) * 16;
        v8f acc = {};
#pragma unroll
        for (int pool = 0; pool < 2; ++pool) {
#pragma unroll
            for (int kb = 0; kb < R_DIM / 32; ++kb) {
                const int k0 = kb * 32;
                v16h a = load_a_frag(hs[pool], HSTRIDE, m, khi, k0);
                v16h b = load_b_frag_wT(w2h, R_DIM, nn, kb16, n0, k0);
                acc = __builtin_amdgcn_wmma_f32_16x16x32_f16(
                    false, a, false, b, (short)0, acc, false, false);
            }
        }
#pragma unroll
        for (int r = 0; r < 8; ++r) {
            float v = acc[r];
            y[(size_t)(g0 + mb + r) * C_DIM + n0 + nn] = 1.f / (1.f + __expf(-v));
        }
    }
}

// ---------------------------------------------------------------------------
// Kernel 3: out[n,c] = x[n,c] * y[batch[n], c]
// float4 (128-bit) accesses; x/out streamed with non-temporal hints so the
// hot 4 MB y table stays resident in L2.
// ---------------------------------------------------------------------------
__global__ __launch_bounds__(256) void apply_kernel(
    const float* __restrict__ x, const int* __restrict__ batch,
    const float* __restrict__ y, float* __restrict__ out, int n4)
{
    int i = blockIdx.x * 256 + threadIdx.x;
    if (i >= n4) return;
    int n  = i >> 6;        // C/4 = 64 float4 per row
    int c4 = i & 63;
    int g  = batch[n];

    v4f xv = __builtin_nontemporal_load(reinterpret_cast<const v4f*>(x) + i);
    v4f yv = reinterpret_cast<const v4f*>(y)[(size_t)g * (C_DIM / 4) + c4];
    v4f o  = xv * yv;
    __builtin_nontemporal_store(o, reinterpret_cast<v4f*>(out) + i);
}

// ---------------------------------------------------------------------------
extern "C" void kernel_launch(void* const* d_in, const int* in_sizes, int n_in,
                              void* d_out, int out_size, void* d_ws, size_t ws_size,
                              hipStream_t stream)
{
    const float* x     = (const float*)d_in[0];
    const int*   batch = (const int*)d_in[1];
    const float* w1    = (const float*)d_in[2];
    const float* w2    = (const float*)d_in[3];
    const int    N     = in_sizes[0] / C_DIM;

    // workspace layout
    char* ws = (char*)d_ws;
    const size_t seg_bytes = (size_t)G_DIM * C_DIM * sizeof(float);   // 4 MB
    float*     segmax = (float*)(ws);
    float*     segsum = (float*)(ws + seg_bytes);
    float*     yG     = (float*)(ws + 2 * seg_bytes);
    _Float16*  w1h    = (_Float16*)(ws + 3 * seg_bytes);
    _Float16*  w2h    = (_Float16*)(ws + 3 * seg_bytes + (size_t)R_DIM * C_DIM * sizeof(_Float16));

    convert_w_kernel<<<(2 * R_DIM * C_DIM) / 256, 256, 0, stream>>>(w1, w2, w1h, w2h);
    segpool_kernel<<<G_DIM, 256, 0, stream>>>(x, batch, N, segmax, segsum);
    mlp_wmma_kernel<<<G_DIM / 16, 256, 0, stream>>>(segmax, segsum, w1h, w2h, yG);

    const int n4 = N * (C_DIM / 4);
    apply_kernel<<<(n4 + 255) / 256, 256, 0, stream>>>(x, batch, yG, (float*)d_out, n4);
}
